// GRU_59279138620002
// MI455X (gfx1250) — matrix operand order repaired and checked by reading, not compiled, not run on hardware
//
#include <hip/hip_runtime.h>
#include <hip/hip_bf16.h>
#include <math.h>
#include <stdint.h>

#define B_ 64
#define S_ 1024
#define I_ 256
#define H_ 1024
#define G_ (3 * H_)

typedef __attribute__((ext_vector_type(16))) __bf16 v16bf;
typedef __attribute__((ext_vector_type(8)))  __bf16 v8bf;
typedef __attribute__((ext_vector_type(8)))  float  v8f;

__device__ __forceinline__ __bf16 f32_to_bf16(float f) {
  union { float f; unsigned u; } v; v.f = f;
  unsigned r = v.u + 0x7FFFu + ((v.u >> 16) & 1u);  // round-to-nearest-even
  unsigned short hs = (unsigned short)(r >> 16);
  __bf16 o; __builtin_memcpy(&o, &hs, 2); return o;
}

__device__ __forceinline__ v8f wmma_bf16(v16bf a, v16bf b, v8f c) {
  // D = A(16x32 bf16) x B(32x16 bf16) + C(16x16 f32)
  return __builtin_amdgcn_wmma_f32_16x16x32_bf16(false, a, false, b, (short)0, c,
                                                 false, false);
}

// Async DMA of 16 bytes per lane: global -> LDS, tracked by ASYNCcnt.
// VDST operand is the LDS byte address (low 32 bits of the flat LDS pointer).
__device__ __forceinline__ void async_ld16(void* lds, const void* gsrc) {
  uint32_t l = (uint32_t)(uintptr_t)lds;
  asm volatile("global_load_async_to_lds_b128 %0, %1, off"
               :: "v"(l), "v"(gsrc) : "memory");
}

__device__ __forceinline__ void wait_async0() {
  asm volatile("s_wait_asynccnt 0x0" ::: "memory");
}

// ---- fragment loaders -------------------------------------------------------
// A-matrix 16x32 bf16 layout (contiguous k, matching B): lanes 0-15 hold M=0..15
// with K=0..15; lanes 16-31 hold K=16..31.
__device__ __forceinline__ v16bf lds_a_frag(const __bf16* tile /*[64][32]*/, int m16) {
  int lane = threadIdx.x & 31;
  int m  = m16 * 16 + (lane & 15);
  int kh = (lane >> 4) * 16;
  v8bf lo = *(const v8bf*)(tile + m * 32 + kh);        // K = kh..kh+7
  v8bf hi = *(const v8bf*)(tile + m * 32 + 8 + kh);    // K = kh+8..kh+15
  v16bf a;
#pragma unroll
  for (int i = 0; i < 8; ++i) { a[i] = lo[i]; a[i + 8] = hi[i]; }
  return a;
}

// B-matrix 32x16 bf16: lane = output column (n), lane[4] selects K 0..15 vs
// 16..31; one column of W^T == one contiguous row of W (row-major), so this is
// a 32-byte contiguous per-lane load.
__device__ __forceinline__ v16bf gbl_b_frag(const __bf16* __restrict__ W,
                                            int row_base, int K, int kk) {
  int lane = threadIdx.x & 31;
  long row = row_base + (lane & 15);
  int  k0  = kk + ((lane >> 4) << 4);
  return *(const v16bf*)(W + row * (long)K + k0);
}

// Stage a 64x32 bf16 A tile into LDS: 256 lanes x 16B async copies.
__device__ __forceinline__ void stage_async(const __bf16* __restrict__ A, __bf16* tile,
                                            int row0, int K, int kk) {
  int t = threadIdx.x;
  int r = t >> 2;
  int c = (t & 3) * 8;
  async_ld16(tile + r * 32 + c, A + (long)(row0 + r) * K + kk + c);
}

// ---- pipelined GEMM core ----------------------------------------------------
// WG = 256 threads = 8 waves; WG tile 64(M) x 64(N) per gate; each wave owns a
// 16x32 strip (2 accumulators) per gate. One barrier per K-step: retire the
// current A-tile DMA, barrier, then immediately launch next-step traffic
// (async A DMA + B fragment loads) so it overlaps the WMMA phase and is only
// waited at the NEXT barrier.
template <int KTOT, int NG>
__device__ __forceinline__ void gemm_core(const __bf16* __restrict__ A, int arow0,
                                          const __bf16* __restrict__ W, int wrow0,
                                          int wstride, __bf16* tiles /*2x64x32*/,
                                          v8f (&acc)[NG][2]) {
  constexpr int NITER = KTOT / 32;
  int w = threadIdx.x >> 5;
  int m16 = w & 3, npair = w >> 2;
  v16bf bc[NG][2], bn[NG][2];

  stage_async(A, tiles, arow0, KTOT, 0);  // prologue: tile 0 in flight
#pragma unroll
  for (int g = 0; g < NG; ++g) {
    int rb = wrow0 + g * wstride + npair * 32;
    bc[g][0] = gbl_b_frag(W, rb,      KTOT, 0);
    bc[g][1] = gbl_b_frag(W, rb + 16, KTOT, 0);
  }

#pragma unroll
  for (int ki = 0; ki < NITER; ++ki) {
    __bf16* cur = tiles + (ki & 1) * (64 * 32);
    __bf16* nxt = tiles + ((ki + 1) & 1) * (64 * 32);
    // Only cur's DMA (issued last iteration) is outstanding: retire it. It had
    // a full compute phase to land.
    wait_async0();
    __syncthreads();  // cur visible everywhere; everyone's reads of nxt drained
    if (ki + 1 < NITER) {
      // Launch next-step traffic AFTER the barrier so the barrier's implied
      // s_wait_loadcnt 0 never stalls on fresh prefetches.
      stage_async(A, nxt, arow0, KTOT, (ki + 1) * 32);
#pragma unroll
      for (int g = 0; g < NG; ++g) {
        int rb = wrow0 + g * wstride + npair * 32;
        bn[g][0] = gbl_b_frag(W, rb,      KTOT, (ki + 1) * 32);
        bn[g][1] = gbl_b_frag(W, rb + 16, KTOT, (ki + 1) * 32);
      }
    }
    v16bf a = lds_a_frag(cur, m16);
#pragma unroll
    for (int g = 0; g < NG; ++g) {
      acc[g][0] = wmma_bf16(a, bc[g][0], acc[g][0]);
      acc[g][1] = wmma_bf16(a, bc[g][1], acc[g][1]);
    }
#pragma unroll
    for (int g = 0; g < NG; ++g) { bc[g][0] = bn[g][0]; bc[g][1] = bn[g][1]; }
  }
}

// ---- helper kernels ---------------------------------------------------------
__global__ __launch_bounds__(256) void cvt_f32_bf16(const float* __restrict__ src,
                                                    __bf16* __restrict__ dst, int n) {
  int i = blockIdx.x * blockDim.x + threadIdx.x;
  int stride = gridDim.x * blockDim.x;
  for (; i < n; i += stride) dst[i] = f32_to_bf16(src[i]);
}

__global__ __launch_bounds__(256) void zero_h(float* __restrict__ hf,
                                              __bf16* __restrict__ hb) {
  int i = blockIdx.x * blockDim.x + threadIdx.x;  // exactly B_*H_ threads
  hf[i] = 0.0f;
  hb[i] = f32_to_bf16(0.0f);
}

// ---- x_gates = seq @ W_ih^T + b_ih : [B*S, 3H] ------------------------------
__global__ __launch_bounds__(256) void xgates_gemm(const __bf16* __restrict__ A,    // [B*S, I]
                                                   const __bf16* __restrict__ Wih,  // [3H, I]
                                                   const float* __restrict__ bih,
                                                   float* __restrict__ xg) {        // [B*S, 3H]
  __shared__ __bf16 tiles[2 * 64 * 32];
  int w = threadIdx.x >> 5, lane = threadIdx.x & 31;
  int m16 = w & 3, npair = w >> 2;
  int row0 = blockIdx.y * 64;
  int col0 = blockIdx.x * 64;
  v8f acc[1][2];
  acc[0][0] = (v8f){}; acc[0][1] = (v8f){};
  gemm_core<I_, 1>(A, row0, Wih, col0, 0, tiles, acc);

  int n0 = col0 + npair * 32 + (lane & 15);
  int n1 = n0 + 16;
  float bb0 = bih[n0], bb1 = bih[n1];
#pragma unroll
  for (int r = 0; r < 8; ++r) {
    int m = row0 + m16 * 16 + r + 8 * (lane >> 4);
    xg[(long)m * G_ + n0] = acc[0][0][r] + bb0;
    xg[(long)m * G_ + n1] = acc[0][1][r] + bb1;
  }
}

// ---- one GRU timestep: hg = h_prev @ W_hh^T, fused gate math ----------------
__global__ __launch_bounds__(256) void gru_step(const float* __restrict__ xg,       // [B*S, 3H]
                                                const __bf16* __restrict__ hprev_bf,// [B, H]
                                                const float* __restrict__ hprev_f,  // [B, H]
                                                const __bf16* __restrict__ Whh,     // [3H, H]
                                                const float* __restrict__ bhh,
                                                float* __restrict__ hnext_f,
                                                __bf16* __restrict__ hnext_bf,
                                                int t) {
  __shared__ __bf16 tiles[2 * 64 * 32];
  int w = threadIdx.x >> 5, lane = threadIdx.x & 31;
  int m16 = w & 3, npair = w >> 2;
  int col0 = blockIdx.x * 64;  // h-column tile; M tile = whole batch (B == 64)
  v8f acc[3][2];
#pragma unroll
  for (int g = 0; g < 3; ++g) { acc[g][0] = (v8f){}; acc[g][1] = (v8f){}; }
  gemm_core<H_, 3>(hprev_bf, 0, Whh, col0, H_, tiles, acc);

#pragma unroll
  for (int j = 0; j < 2; ++j) {
    int n = col0 + npair * 32 + j * 16 + (lane & 15);
    float br = bhh[n], bz = bhh[H_ + n], bn = bhh[2 * H_ + n];
#pragma unroll
    for (int r = 0; r < 8; ++r) {
      int m = m16 * 16 + r + 8 * (lane >> 4);  // batch index
      const float* xrow = xg + ((long)m * S_ + t) * G_;
      float rr = 1.0f / (1.0f + __expf(-(xrow[n]          + acc[0][j][r] + br)));
      float zz = 1.0f / (1.0f + __expf(-(xrow[H_ + n]     + acc[1][j][r] + bz)));
      float nn = tanhf(xrow[2 * H_ + n] + rr * (acc[2][j][r] + bn));
      float hp = hprev_f[m * H_ + n];
      float hv = (1.0f - zz) * nn + zz * hp;
      hnext_f[m * H_ + n]  = hv;
      hnext_bf[m * H_ + n] = f32_to_bf16(hv);
    }
  }
}

// ---- out = h_last @ W_out^T + b_out : [B, I] --------------------------------
__global__ __launch_bounds__(256) void head_gemm(const __bf16* __restrict__ hbf,   // [B, H]
                                                 const __bf16* __restrict__ Wout,  // [I, H]
                                                 const float* __restrict__ bout,
                                                 float* __restrict__ out) {        // [B, I]
  __shared__ __bf16 tiles[2 * 64 * 32];
  int w = threadIdx.x >> 5, lane = threadIdx.x & 31;
  int m16 = w & 3, npair = w >> 2;
  int col0 = blockIdx.x * 64;
  v8f acc[1][2];
  acc[0][0] = (v8f){}; acc[0][1] = (v8f){};
  gemm_core<H_, 1>(hbf, 0, Wout, col0, 0, tiles, acc);

  int n0 = col0 + npair * 32 + (lane & 15);
  int n1 = n0 + 16;
  float bb0 = bout[n0], bb1 = bout[n1];
#pragma unroll
  for (int r = 0; r < 8; ++r) {
    int m = m16 * 16 + r + 8 * (lane >> 4);
    out[m * I_ + n0] = acc[0][0][r] + bb0;
    out[m * I_ + n1] = acc[0][1][r] + bb1;
  }
}

// ---- launcher ---------------------------------------------------------------
extern "C" void kernel_launch(void* const* d_in, const int* in_sizes, int n_in,
                              void* d_out, int out_size, void* d_ws, size_t ws_size,
                              hipStream_t stream) {
  const float* seq  = (const float*)d_in[0];
  const float* Wih  = (const float*)d_in[1];
  const float* Whh  = (const float*)d_in[2];
  const float* bih  = (const float*)d_in[3];
  const float* bhh  = (const float*)d_in[4];
  const float* Wout = (const float*)d_in[5];
  const float* bout = (const float*)d_in[6];

  char* ws = (char*)d_ws;
  size_t off = 0;
  auto alloc = [&](size_t bytes) -> void* {
    void* p = ws + off;
    off = (off + bytes + 255) & ~(size_t)255;
    return p;
  };
  __bf16* seq_bf  = (__bf16*)alloc((size_t)B_ * S_ * I_ * 2);
  __bf16* wih_bf  = (__bf16*)alloc((size_t)G_ * I_ * 2);
  __bf16* whh_bf  = (__bf16*)alloc((size_t)G_ * H_ * 2);
  __bf16* wout_bf = (__bf16*)alloc((size_t)I_ * H_ * 2);
  float*  xg      = (float*) alloc((size_t)B_ * S_ * G_ * 4);
  float*  hf0     = (float*) alloc((size_t)B_ * H_ * 4);
  float*  hf1     = (float*) alloc((size_t)B_ * H_ * 4);
  __bf16* hb0     = (__bf16*)alloc((size_t)B_ * H_ * 2);
  __bf16* hb1     = (__bf16*)alloc((size_t)B_ * H_ * 2);

  // Precision staging: fp32 -> bf16 operands for WMMA.
  cvt_f32_bf16<<<2048, 256, 0, stream>>>(seq,  seq_bf,  B_ * S_ * I_);
  cvt_f32_bf16<<<256,  256, 0, stream>>>(Wih,  wih_bf,  G_ * I_);
  cvt_f32_bf16<<<512,  256, 0, stream>>>(Whh,  whh_bf,  G_ * H_);
  cvt_f32_bf16<<<256,  256, 0, stream>>>(Wout, wout_bf, I_ * H_);
  zero_h<<<(B_ * H_) / 256, 256, 0, stream>>>(hf0, hb0);

  // Bulk input projection: [65536, 256] x [256, 3072].
  xgates_gemm<<<dim3(G_ / 64, (B_ * S_) / 64), 256, 0, stream>>>(seq_bf, wih_bf, bih, xg);

  // Serial recurrence: 1024 fused GEMM+gate steps, h double-buffered.
  float*  hf[2] = {hf0, hf1};
  __bf16* hb[2] = {hb0, hb1};
  for (int t = 0; t < S_; ++t) {
    gru_step<<<H_ / 64, 256, 0, stream>>>(xg, hb[t & 1], hf[t & 1], whh_bf, bhh,
                                          hf[(t + 1) & 1], hb[(t + 1) & 1], t);
  }

  // Final h lands in buffer 0 (t=1023 writes (1023+1)&1 == 0).
  head_gemm<<<I_ / 64, 256, 0, stream>>>(hb[0], wout_bf, bout, (float*)d_out);
}
